// GraphSAGE_30374008717351
// MI455X (gfx1250) — compile-verified, matrix operand order checked by reading, not run on hardware
//
#include <hip/hip_runtime.h>
#include <hip/hip_bf16.h>
#include <math.h>

// ---------------------------------------------------------------------------
// GraphSAGE (3 layers) on MI455X / gfx1250.
//   layer: aggr = segment_mean(x[src]*w, dst);  out = [aggr|x] @ [Wl;Wr] + b
// Scatter phase: f32 global atomics (25.6MB accumulator lives in 192MB L2).
// GEMM phase: bf16 WMMA v_wmma_f32_16x16x32_bf16, f32 accumulate.
//   - packed weights staged once per block into LDS (<=64KB of the WGP's
//     320KB), inner loop feeds WMMA from ds_load_b128 (low latency, no L0
//     pressure; the vmem pipe is left to the streaming A reads).
//   - 50000 rows = 3125 row-tiles = 625 blocks x 5 waves exactly: no row
//     guards, EXEC stays all-ones through the WMMA pipeline.
// ---------------------------------------------------------------------------

typedef __attribute__((ext_vector_type(16))) __bf16 v16bf;
typedef __attribute__((ext_vector_type(8)))  __bf16 v8bf;
typedef __attribute__((ext_vector_type(8)))  float  v8f;

#define N_NODES 50000
#define N_EDGES 800000
#define F_IN    128
#define HID     128
#define N_CLS   40
#define KA      256          // concat K = 2*128
#define KTILES  8            // 256 / 32
#define GEMM_WAVES 5         // 5 waves/block -> 3125 tiles / 5 = 625 blocks
#define FRAG_BF16 (32 * 16)  // bf16 elements per 32x16 B fragment

// ------------------------------ zero fill ----------------------------------
__global__ void sage_zero_f32(float* __restrict__ p, size_t n) {
    size_t i = (size_t)blockIdx.x * blockDim.x + threadIdx.x;
    size_t stride = (size_t)gridDim.x * blockDim.x;
    for (; i < n; i += stride) p[i] = 0.0f;
}

// --------------------------- edge scatter-add ------------------------------
// 32 lanes per edge; each lane handles features lane, lane+32, lane+64, lane+96
// -> fully coalesced 128B gathers per quarter-row, atomics land in L2.
__global__ __launch_bounds__(256)
void sage_scatter(const float* __restrict__ x,
                  const long long* __restrict__ ei,   // [2][E] int64
                  const float* __restrict__ ew,
                  float* __restrict__ summed,          // [N][128]
                  float* __restrict__ deg,             // [N]
                  int nEdges, int computeDeg) {
    int t = blockIdx.x * 256 + threadIdx.x;
    int e = t >> 5;
    if (e >= nEdges) return;
    int lane = t & 31;
    long long s = ei[e];
    long long d = ei[(size_t)nEdges + e];
    float w = ew[e];
    const float* xs = x + (size_t)s * 128;
    float* sd = summed + (size_t)d * 128;
#pragma unroll
    for (int i = 0; i < 4; ++i) {
        int f = lane + i * 32;
        atomicAdd(sd + f, xs[f] * w);
    }
    if (computeDeg && lane == 0) atomicAdd(deg + d, 1.0f);
}

// --------------------- finalize mean + build bf16 A ------------------------
// abuf row n = [ aggr(n) | x(n) ]  as bf16, 256 elements (512B, 16B aligned).
__global__ __launch_bounds__(256)
void sage_finalize(const float* __restrict__ summed,
                   const float* __restrict__ deg,
                   const float* __restrict__ xin,
                   __bf16* __restrict__ abuf, int nNodes) {
    int t = blockIdx.x * 256 + threadIdx.x;
    if (t >= nNodes * 128) return;
    int n = t >> 7, f = t & 127;
    float inv = 1.0f / fmaxf(deg[n], 1.0f);
    abuf[(size_t)n * KA + f]        = (__bf16)(summed[t] * inv);
    abuf[(size_t)n * KA + 128 + f]  = (__bf16)(xin[t]);
}

// ----------------------- weight repack to B fragments ----------------------
// Fragment layout matches ISA 16-bit B operand (32x16 tile):
//   lane = half*16 + idx ; element e -> K = kt*32 + (e<8 ? half*8+e
//                                                       : 16+half*8+(e-8))
//   col = ct*16 + idx
// Stored so each lane reads one contiguous v16bf (32B).
__global__ __launch_bounds__(256)
void sage_repack_w(const float* __restrict__ Wl,   // [128][nCols] row-major
                   const float* __restrict__ Wr,   // [128][nCols]
                   int nCols, __bf16* __restrict__ Bp, int nColTiles) {
    int t = blockIdx.x * 256 + threadIdx.x;
    int total = nColTiles * KTILES * 32;
    if (t >= total) return;
    int lane = t & 31;
    int kt   = (t >> 5) & (KTILES - 1);
    int ct   = t >> 8;
    int half = lane >> 4, idx = lane & 15;
    int col  = ct * 16 + idx;
    __bf16* dst = Bp + (size_t)t * 16;
#pragma unroll
    for (int e = 0; e < 16; ++e) {
        int k = kt * 32 + ((e < 8) ? (half * 8 + e) : (16 + half * 8 + (e - 8)));
        float v = 0.0f;
        if (col < nCols)
            v = (k < 128) ? Wl[k * nCols + col] : Wr[(k - 128) * nCols + col];
        dst[e] = (__bf16)v;
    }
}

// ------------------------------- WMMA GEMM ---------------------------------
// out[N][ldOut] = relu?( A[N][256]_bf16 @ W_packed + bias )
// Block = 160 threads = 5 waves. Packed weights for the whole layer are
// staged into LDS once per block (nColTiles*8KB), then each wave owns one
// 16-row tile, reusing its 8 register-resident A fragments while streaming
// B fragments from LDS (ds_load_b128).
__global__ __launch_bounds__(32 * GEMM_WAVES)
void sage_gemm_wmma(const __bf16* __restrict__ A,
                    const __bf16* __restrict__ Bp,
                    const float* __restrict__ bias, int biasLen,
                    float* __restrict__ out, int ldOut,
                    int nColTiles, int doRelu) {
    extern __shared__ __bf16 Bs[];   // nColTiles * KTILES * FRAG_BF16 bf16

    const int tid  = threadIdx.x;
    const int wave = tid >> 5;
    const int lane = tid & 31;
    const int half = lane >> 4;
    const int idx  = lane & 15;
    const int rowTile = blockIdx.x * GEMM_WAVES + wave;   // always < 3125
    const int arow = rowTile * 16 + idx;                  // row this lane holds

    // ---- cooperative stage of packed weights into LDS (16B chunks)
    {
        const int nChunks = nColTiles * KTILES * FRAG_BF16 / 8;  // uint4 count
        const uint4* src = (const uint4*)Bp;
        uint4* dst = (uint4*)Bs;
        for (int i = tid; i < nChunks; i += 32 * GEMM_WAVES)
            dst[i] = src[i];
    }

    // ---- load all 8 A fragments (K = 256) once; ISA 16-bit A layout:
    //   lanes 0-15  : K = kt*32 + [0..7]   and  kt*32 + [16..23]
    //   lanes 16-31 : K = kt*32 + [8..15]  and  kt*32 + [24..31]
    const __bf16* aptr = A + (size_t)arow * KA;
    v16bf afrag[KTILES];
#pragma unroll
    for (int kt = 0; kt < KTILES; ++kt) {
        v8bf lo = *(const v8bf*)(aptr + kt * 32 + half * 8);        // 16B load
        v8bf hi = *(const v8bf*)(aptr + kt * 32 + 16 + half * 8);   // 16B load
        afrag[kt] = __builtin_shufflevector(lo, hi,
                        0,1,2,3,4,5,6,7,8,9,10,11,12,13,14,15);
    }

    __syncthreads();

    for (int ct = 0; ct < nColTiles; ++ct) {
        const __bf16* bsp = Bs + ((size_t)ct * KTILES * 32 + lane) * 16;

        v8f acc = {};
#pragma unroll
        for (int kt = 0; kt < KTILES; ++kt) {
            v16bf bfrag = *(const v16bf*)(bsp + (size_t)kt * FRAG_BF16); // 2x ds_load_b128
            acc = __builtin_amdgcn_wmma_f32_16x16x32_bf16(
                      /*neg_a=*/false, afrag[kt],
                      /*neg_b=*/false, bfrag,
                      /*c_mod=*/(short)0, acc,
                      /*reuse_a=*/false, /*reuse_b=*/false);
        }

        // ---- epilogue: C/D layout VGPR r -> M = r + half*8, N = idx
        int col = ct * 16 + idx;
        float bv = (col < biasLen) ? bias[col] : 0.0f;
        float* orow0 = out + (size_t)(rowTile * 16 + half * 8) * ldOut + col;
#pragma unroll
        for (int r = 0; r < 8; ++r) {
            float v = acc[r] + bv;
            if (doRelu) v = fmaxf(v, 0.0f);
            orow0[(size_t)r * ldOut] = v;
        }
    }
}

// ------------------------------ log_softmax --------------------------------
__global__ __launch_bounds__(256)
void sage_logsoftmax(const float* __restrict__ h, float* __restrict__ out,
                     int nNodes) {
    int n = blockIdx.x * 256 + threadIdx.x;
    if (n >= nNodes) return;
    const float* r = h + (size_t)n * 48;
    float m = -INFINITY;
#pragma unroll
    for (int c = 0; c < N_CLS; ++c) m = fmaxf(m, r[c]);
    float s = 0.0f;
#pragma unroll
    for (int c = 0; c < N_CLS; ++c) s += __expf(r[c] - m);
    float ls = __logf(s) + m;
    float* o = out + (size_t)n * N_CLS;
#pragma unroll
    for (int c = 0; c < N_CLS; ++c) o[c] = r[c] - ls;
}

// ---------------------------------------------------------------------------
static inline size_t alignUp(size_t v, size_t a) { return (v + a - 1) & ~(a - 1); }

extern "C" void kernel_launch(void* const* d_in, const int* in_sizes, int n_in,
                              void* d_out, int out_size, void* d_ws, size_t ws_size,
                              hipStream_t stream) {
    (void)in_sizes; (void)n_in; (void)out_size; (void)ws_size;
    const float*     x   = (const float*)d_in[0];
    const long long* ei  = (const long long*)d_in[1];   // int64 [2][E]
    const float*     ew  = (const float*)d_in[2];
    const float*     Wl1 = (const float*)d_in[3];
    const float*     bl1 = (const float*)d_in[4];
    const float*     Wr1 = (const float*)d_in[5];
    const float*     Wl2 = (const float*)d_in[6];
    const float*     bl2 = (const float*)d_in[7];
    const float*     Wr2 = (const float*)d_in[8];
    const float*     Wl3 = (const float*)d_in[9];
    const float*     bl3 = (const float*)d_in[10];
    const float*     Wr3 = (const float*)d_in[11];
    float* outp = (float*)d_out;

    // ---- carve workspace
    char* ws = (char*)d_ws;
    size_t off = 0;
    float*  deg    = (float*)(ws + off); off = alignUp(off + (size_t)N_NODES * 4, 256);
    float*  summed = (float*)(ws + off); off = alignUp(off + (size_t)N_NODES * 128 * 4, 256);
    __bf16* abuf   = (__bf16*)(ws + off); off = alignUp(off + (size_t)N_NODES * KA * 2, 256);
    __bf16* bpack  = (__bf16*)(ws + off); off = alignUp(off + (size_t)8 * KTILES * FRAG_BF16 * 2, 256);
    float*  h1     = (float*)(ws + off); off = alignUp(off + (size_t)N_NODES * 128 * 4, 256);
    float*  h2     = (float*)(ws + off); off = alignUp(off + (size_t)N_NODES * 128 * 4, 256);
    float*  h3     = (float*)(ws + off); off = alignUp(off + (size_t)N_NODES * 48 * 4, 256);

    const int zeroBlocks    = 2048;
    const int scatterBlocks = (N_EDGES * 32 + 255) / 256;
    const int finBlocks     = (N_NODES * 128 + 255) / 256;
    const int gemmBlocks    = (N_NODES / 16) / GEMM_WAVES;      // 3125/5 = 625
    const int lsmBlocks     = (N_NODES + 255) / 256;
    const size_t lds8 = (size_t)8 * KTILES * FRAG_BF16 * 2;     // 64KB
    const size_t lds3 = (size_t)3 * KTILES * FRAG_BF16 * 2;     // 24KB

    // =================== layer 1: x -> h1 (relu) ===================
    sage_zero_f32<<<zeroBlocks, 256, 0, stream>>>(summed, (size_t)N_NODES * 128);
    sage_zero_f32<<<zeroBlocks, 256, 0, stream>>>(deg, (size_t)N_NODES);
    sage_scatter<<<scatterBlocks, 256, 0, stream>>>(x, ei, ew, summed, deg, N_EDGES, 1);
    sage_finalize<<<finBlocks, 256, 0, stream>>>(summed, deg, x, abuf, N_NODES);
    sage_repack_w<<<(8 * KTILES * 32 + 255) / 256, 256, 0, stream>>>(Wl1, Wr1, HID, bpack, 8);
    sage_gemm_wmma<<<gemmBlocks, 32 * GEMM_WAVES, lds8, stream>>>(abuf, bpack, bl1, HID,
                                                                  h1, HID, 8, 1);

    // =================== layer 2: h1 -> h2 (relu) ===================
    sage_zero_f32<<<zeroBlocks, 256, 0, stream>>>(summed, (size_t)N_NODES * 128);
    sage_scatter<<<scatterBlocks, 256, 0, stream>>>(h1, ei, ew, summed, deg, N_EDGES, 0);
    sage_finalize<<<finBlocks, 256, 0, stream>>>(summed, deg, h1, abuf, N_NODES);
    sage_repack_w<<<(8 * KTILES * 32 + 255) / 256, 256, 0, stream>>>(Wl2, Wr2, HID, bpack, 8);
    sage_gemm_wmma<<<gemmBlocks, 32 * GEMM_WAVES, lds8, stream>>>(abuf, bpack, bl2, HID,
                                                                  h2, HID, 8, 1);

    // ============ layer 3: h2 -> h3 (40 cols padded to 48) ============
    sage_zero_f32<<<zeroBlocks, 256, 0, stream>>>(summed, (size_t)N_NODES * 128);
    sage_scatter<<<scatterBlocks, 256, 0, stream>>>(h2, ei, ew, summed, deg, N_EDGES, 0);
    sage_finalize<<<finBlocks, 256, 0, stream>>>(summed, deg, h2, abuf, N_NODES);
    sage_repack_w<<<(3 * KTILES * 32 + 255) / 256, 256, 0, stream>>>(Wl3, Wr3, N_CLS, bpack, 3);
    sage_gemm_wmma<<<gemmBlocks, 32 * GEMM_WAVES, lds3, stream>>>(abuf, bpack, bl3, N_CLS,
                                                                  h3, 48, 3, 0);

    // ======================== log_softmax ========================
    sage_logsoftmax<<<lsmBlocks, 256, 0, stream>>>(h3, outp, N_NODES);
}
